// PointCloud3DCNN_58858231824589
// MI455X (gfx1250) — matrix operand order repaired and checked by reading, not compiled, not run on hardware
//
#include <hip/hip_runtime.h>
#include <hip/hip_bf16.h>

typedef __attribute__((ext_vector_type(16))) _Float16 v16h;
typedef __attribute__((ext_vector_type(8)))  _Float16 v8h;
typedef __attribute__((ext_vector_type(2)))  _Float16 h2;
typedef __attribute__((ext_vector_type(8)))  float    v8f;

#define AS1 __attribute__((address_space(1)))
#define AS3 __attribute__((address_space(3)))

#if defined(__gfx1250__) && __has_builtin(__builtin_amdgcn_global_load_async_to_lds_b32)
#define HAVE_ASYNC_LDS 1
#else
#define HAVE_ASYNC_LDS 0
#endif

static __device__ __forceinline__ void wait_async0() {
#if defined(__gfx1250__)
#if __has_builtin(__builtin_amdgcn_s_wait_asynccnt)
    __builtin_amdgcn_s_wait_asynccnt(0);
#else
    asm volatile("s_wait_asynccnt 0x0" ::: "memory");
#endif
#endif
}

// ---------------- conv modes ----------------
enum : int { CONV_S2 = 0, CONV_1X1 = 1, CONVT_S2 = 2 };

// Branchless im2col address: returns clamped flat index + validity.
// tap < 32  ->  tap/9 == (tap*57)>>9 ;  tr < 9 -> tr/3 == (tr*11)>>5 (exact).
template<int MODE>
static __device__ __forceinline__
void addrA(int b, int zo, int yo, int xo, int kk,
           int K, int Cin, int lgCin, int Gi, int lgGi,
           int& idx, bool& ok)
{
    int cin, zi, yi, xi;
    if (MODE == CONV_1X1) {
        cin = kk; zi = zo; yi = yo; xi = xo;
        ok = (kk < K) && (cin < Cin);
    } else {
        int tap = (kk >> lgCin) & 31;
        cin = kk & ((1 << lgCin) - 1);
        int tz = (tap * 57) >> 9;  int tr = tap - tz * 9;
        int ty = (tr * 11) >> 5;   int tx = tr - ty * 3;
        if (MODE == CONV_S2) {
            // jax SAME, stride 2, k=3: pad_lo = 0, pad_hi = 1
            zi = 2 * zo + tz; yi = 2 * yo + ty; xi = 2 * xo + tx;
            ok = (kk < K) && (cin < Cin) && (zi < Gi) && (yi < Gi) && (xi < Gi);
        } else {  // CONVT_S2: o = 2*i + k - 1 -> parity-gated gather
            int z2 = zo + 1 - tz, y2 = yo + 1 - ty, x2 = xo + 1 - tx;
            ok = (kk < K) && (cin < Cin) && !((z2 | y2 | x2) & 1) &&
                 (z2 >= 0) && (y2 >= 0) && (x2 >= 0);
            zi = z2 >> 1; yi = y2 >> 1; xi = x2 >> 1;
            ok = ok && (zi < Gi) && (yi < Gi) && (xi < Gi);
        }
    }
    int vox = (((zi << lgGi) + yi) << lgGi) + xi;
    idx = ok ? ((b * Cin + cin) * (1 << (3 * lgGi)) + vox) : 0;
}

// ============================================================
// One-time weight repack: f32 [tap][Cin][Cout] -> f16 column-major
// [Coutp][Kpad] with zero padding baked in.  kk = tap*Cinp + cin.
// ============================================================
__global__ void repack_w_kernel(const float* __restrict__ w, _Float16* __restrict__ bp,
                                int taps, int Cin, int lgCin, int Cout,
                                int Kpad, int Coutp)
{
    int idx = blockIdx.x * blockDim.x + threadIdx.x;
    if (idx >= Coutp * Kpad) return;
    int n  = idx / Kpad;
    int kk = idx - n * Kpad;
    int tap = kk >> lgCin;
    int cin = kk & ((1 << lgCin) - 1);
    float v = (tap < taps && cin < Cin && n < Cout)
            ? w[(size_t)(tap * Cin + cin) * Cout + n] : 0.f;
    bp[(size_t)n * Kpad + kk] = (_Float16)v;
}

// ============================================================
// Implicit-GEMM conv via V_WMMA_F32_16X16X32_F16.
//  A tile: 128 x 32 f16, row stride 40 halves (80B, 20-bank skew)
//          -> each lane's fragment = two aligned ds_load_b128.
//          Gather is branchless; loads batched before cvt/store.
//  B tile: 16 cols x 32 halves of the pre-packed f16 weights,
//          one unconditional global_load_async_to_lds_b32 per thread.
// ============================================================
template<int MODE>
__global__ __launch_bounds__(256)
void wmma_conv_kernel(const float* __restrict__ src,     // [B, Cin, Gi^3] (pre-masked for CONVT)
                      const _Float16* __restrict__ bp,   // packed [Coutp][Kpad]
                      float* __restrict__ dst,           // [B, Cout, Go^3]
                      int B, int Cin, int lgCin, int Cout,
                      int Gi, int lgGi, int Go, int lgGo, int Kpad)
{
    const int taps = (MODE == CONV_1X1) ? 1 : 27;
    const int K    = taps << lgCin;          // padded K (cin >= Cin zero-filled)
    const int lg3  = 3 * lgGo;
    const int go3  = 1 << lg3;

    __shared__ _Float16 As[128 * 40] __attribute__((aligned(16)));
    __shared__ _Float16 Bt[16 * 40]  __attribute__((aligned(16)));

    const int tid   = threadIdx.x;
    const int wave  = tid >> 5;
    const int lane  = tid & 31;
    const int mBase = blockIdx.x * 128;
    const int nBase = blockIdx.y * 16;

    if (tid == 0) __builtin_prefetch(bp, 0, 1);

    // per-thread constants for the B-tile copy (address linear in k0)
    const int bcol = tid >> 4;               // 0..15
    const int bhp  = (tid & 15) << 1;        // half-pair 0,2,..,30
    const _Float16* bsrc0 = bp + (size_t)(nBase + bcol) * Kpad + bhp;
    _Float16* bdst = Bt + bcol * 40 + bhp;

    v8f acc = {0.f, 0.f, 0.f, 0.f, 0.f, 0.f, 0.f, 0.f};

    for (int k0 = 0; k0 < Kpad; k0 += 32) {
        // ---- B tile: issue async copy (no masks, no address math) ----
#if HAVE_ASYNC_LDS
        __builtin_amdgcn_global_load_async_to_lds_b32(
            (AS1 int*)(bsrc0 + k0), (AS3 int*)bdst, 0, 0);
#else
        *(h2*)bdst = *(const h2*)(bsrc0 + k0);
#endif

        // ---- A tile phase 1: batched branchless gather (16 loads) ----
        float va[16];
        #pragma unroll
        for (int i = 0; i < 8; ++i) {
            int p  = tid + i * 256;        // pair index, 2048 pairs
            int r  = p >> 4;               // row 0..127
            int kp = (p & 15) << 1;        // K col pair
            int m  = mBase + r;            // M is always a multiple of 128
            int b  = m >> lg3;
            int rv = m & (go3 - 1);
            int zo = rv >> (2 * lgGo);
            int yo = (rv >> lgGo) & (Go - 1);
            int xo = rv & (Go - 1);
            int i0, i1; bool o0, o1;
            addrA<MODE>(b, zo, yo, xo, k0 + kp,     K, Cin, lgCin, Gi, lgGi, i0, o0);
            addrA<MODE>(b, zo, yo, xo, k0 + kp + 1, K, Cin, lgCin, Gi, lgGi, i1, o1);
            float t0 = src[i0];
            float t1 = src[i1];
            va[2 * i]     = o0 ? t0 : 0.f;
            va[2 * i + 1] = o1 ? t1 : 0.f;
        }
        // ---- A tile phase 2: convert + paired LDS stores ----
        #pragma unroll
        for (int i = 0; i < 8; ++i) {
            int p  = tid + i * 256;
            int r  = p >> 4;
            int kp = (p & 15) << 1;
            h2 hv; hv[0] = (_Float16)va[2 * i]; hv[1] = (_Float16)va[2 * i + 1];
            *(h2*)(As + r * 40 + kp) = hv;
        }
#if HAVE_ASYNC_LDS
        wait_async0();
#endif
        __syncthreads();

        // ---- fragments (ISA 7.12.2 layouts): 2x b128 each ----
        v16h a, bf;
        {
            int row = (wave << 4) + (lane & 15);
            int kh  = (lane >> 4) << 3;                 // 0 or 8
            v8h lo = *(const v8h*)(As + row * 40 + kh);
            v8h hi = *(const v8h*)(As + row * 40 + 16 + kh);
            a = __builtin_shufflevector(lo, hi, 0, 1, 2, 3, 4, 5, 6, 7,
                                        8, 9, 10, 11, 12, 13, 14, 15);
        }
        {
            int col = lane & 15;
            int kb2 = (lane >> 4) << 3;                 // 0 or 8
            v8h lo = *(const v8h*)(Bt + col * 40 + kb2);
            v8h hi = *(const v8h*)(Bt + col * 40 + 16 + kb2);
            bf = __builtin_shufflevector(lo, hi, 0, 1, 2, 3, 4, 5, 6, 7,
                                         8, 9, 10, 11, 12, 13, 14, 15);
        }
        acc = __builtin_amdgcn_wmma_f32_16x16x32_f16(false, a, false, bf,
                                                     (short)0, acc, false, false);
        __syncthreads();
    }

    // ---- store D tile ----
    {
        int col   = lane & 15;
        int mhalf = (lane >> 4) << 3;
        int n = nBase + col;
        if (n < Cout) {
            #pragma unroll
            for (int r = 0; r < 8; ++r) {
                int m  = mBase + (wave << 4) + mhalf + r;
                int b  = m >> lg3;
                int rv = m & (go3 - 1);
                dst[(size_t)(b * Cout + n) * go3 + rv] = acc[r];
            }
        }
    }
}

// ============================================================
// BatchNorm statistics: one block per channel.
// ============================================================
__global__ __launch_bounds__(256)
void bn_stats_kernel(const float* __restrict__ x, float* __restrict__ stats,
                     int C, int B, int S)
{
    int c = blockIdx.x;
    float s = 0.f, s2 = 0.f;
    for (int b = 0; b < B; ++b) {
        const float* p = x + (size_t)(b * C + c) * S;
        for (int i = threadIdx.x; i < S; i += 256) {
            float v = p[i]; s += v; s2 += v * v;
        }
    }
    __shared__ float rs[256], rq[256];
    rs[threadIdx.x] = s; rq[threadIdx.x] = s2;
    __syncthreads();
    for (int o = 128; o > 0; o >>= 1) {
        if (threadIdx.x < o) { rs[threadIdx.x] += rs[threadIdx.x + o];
                               rq[threadIdx.x] += rq[threadIdx.x + o]; }
        __syncthreads();
    }
    if (threadIdx.x == 0) {
        float inv = 1.f / (float)(B * S);
        float m = rs[0] * inv;
        stats[c]     = m;
        stats[C + c] = rq[0] * inv - m * m;
    }
}

__global__ void bn_apply_kernel(float* __restrict__ x, const float* __restrict__ stats,
                                const float* __restrict__ g, const float* __restrict__ bta,
                                const float* __restrict__ occ, int C, int B, int lgS)
{
    int S = 1 << lgS;
    int idx = blockIdx.x * blockDim.x + threadIdx.x;
    if (idx >= ((B * C) << lgS)) return;
    int s  = idx & (S - 1);
    int bc = idx >> lgS;
    int c  = bc % C;
    int b  = bc / C;
    float m = stats[c], var = stats[C + c];
    float v = (x[idx] - m) * rsqrtf(var + 1e-5f) * g[c] + bta[c];
    v = fmaxf(v, 0.f);
    if (occ) v *= occ[(b << lgS) + s];
    x[idx] = v;
}

// ============================================================
// occupancy propagation (pow2 grids -> shift/and addressing)
// ============================================================
__global__ void occ_down_kernel(const float* __restrict__ o, float* __restrict__ out,
                                int B, int Gi, int lgGi, int Go, int lgGo)
{
    int idx = blockIdx.x * blockDim.x + threadIdx.x;
    int go3 = 1 << (3 * lgGo);
    if (idx >= (B << (3 * lgGo))) return;
    int b = idx >> (3 * lgGo), rv = idx & (go3 - 1);
    int zo = rv >> (2 * lgGo), yo = (rv >> lgGo) & (Go - 1), xo = rv & (Go - 1);
    const float* p = o + ((size_t)b << (3 * lgGi));
    float any = 0.f;
    for (int tz = 0; tz < 3; ++tz) { int zi = 2 * zo + tz; if (zi >= Gi) continue;
      for (int ty = 0; ty < 3; ++ty) { int yi = 2 * yo + ty; if (yi >= Gi) continue;
        for (int tx = 0; tx < 3; ++tx) { int xi = 2 * xo + tx; if (xi >= Gi) continue;
            if (p[(((zi << lgGi) + yi) << lgGi) + xi] > 0.f) any = 1.f; } } }
    out[idx] = any;
}

__global__ void occ_up_kernel(const float* __restrict__ keep, float* __restrict__ out,
                              int B, int Gi, int lgGi, int Go, int lgGo)   // Go = 2*Gi
{
    int idx = blockIdx.x * blockDim.x + threadIdx.x;
    int go3 = 1 << (3 * lgGo);
    if (idx >= (B << (3 * lgGo))) return;
    int b = idx >> (3 * lgGo), rv = idx & (go3 - 1);
    int zo = rv >> (2 * lgGo), yo = (rv >> lgGo) & (Go - 1), xo = rv & (Go - 1);
    const float* p = keep + ((size_t)b << (3 * lgGi));
    float any = 0.f;
    for (int tz = 0; tz < 3; ++tz) { int z2 = zo + 1 - tz; if (z2 < 0 || (z2 & 1)) continue;
      int zi = z2 >> 1; if (zi >= Gi) continue;
      for (int ty = 0; ty < 3; ++ty) { int y2 = yo + 1 - ty; if (y2 < 0 || (y2 & 1)) continue;
        int yi = y2 >> 1; if (yi >= Gi) continue;
        for (int tx = 0; tx < 3; ++tx) { int x2 = xo + 1 - tx; if (x2 < 0 || (x2 & 1)) continue;
          int xi = x2 >> 1; if (xi >= Gi) continue;
          if (p[(((zi << lgGi) + yi) << lgGi) + xi] > 0.f) any = 1.f; } } }
    out[idx] = any;
}

// ============================================================
// elementwise glue
// ============================================================
__global__ void mul_mask_kernel(float* __restrict__ out, const float* __restrict__ a,
                                const float* __restrict__ occ, int C, int B, int lgS)
{
    int S = 1 << lgS;
    int idx = blockIdx.x * blockDim.x + threadIdx.x;
    if (idx >= ((B * C) << lgS)) return;
    int s = idx & (S - 1), b = (idx >> lgS) / C;
    out[idx] = a[idx] * occ[(b << lgS) + s];
}
__global__ void add_kernel(float* __restrict__ o, const float* __restrict__ a,
                           const float* __restrict__ b, int n)
{ int i = blockIdx.x * blockDim.x + threadIdx.x; if (i < n) o[i] = a[i] + b[i]; }

__global__ void max_kernel(float* __restrict__ o, const float* __restrict__ a,
                           const float* __restrict__ b, int n)
{ int i = blockIdx.x * blockDim.x + threadIdx.x; if (i < n) o[i] = fmaxf(a[i], b[i]); }

__global__ void sigmoid_bias_kernel(float* __restrict__ o, const float* __restrict__ x,
                                    const float* __restrict__ bias, int n)
{ int i = blockIdx.x * blockDim.x + threadIdx.x;
  if (i < n) o[i] = 1.f / (1.f + expf(-(x[i] + bias[0]))); }

__global__ void keep_kernel(float* __restrict__ o, const float* __restrict__ prob,
                            const float* __restrict__ co, int n)
{ int i = blockIdx.x * blockDim.x + threadIdx.x;
  if (i < n) o[i] = (prob[i] > 0.5f && co[i] > 0.f) ? 1.f : 0.f; }

__global__ void gt0_kernel(float* __restrict__ o, const float* __restrict__ a, int n)
{ int i = blockIdx.x * blockDim.x + threadIdx.x; if (i < n) o[i] = (a[i] > 0.f) ? 1.f : 0.f; }

__global__ void sigmoid_mask_kernel(float* __restrict__ x, const float* __restrict__ occ,
                                    int C, int B, int lgS)
{
    int S = 1 << lgS;
    int idx = blockIdx.x * blockDim.x + threadIdx.x;
    if (idx >= ((B * C) << lgS)) return;
    int s = idx & (S - 1), b = (idx >> lgS) / C;
    x[idx] = (1.f / (1.f + expf(-x[idx]))) * occ[(b << lgS) + s];
}
__global__ void addbias_kernel(float* __restrict__ x, const float* __restrict__ bias, int n)
{ int i = blockIdx.x * blockDim.x + threadIdx.x; if (i < n) x[i] += bias[0]; }

// preds postprocess: pyramid (B,12,G^3) -> (B, G^3*3, 3)
__global__ void preds_kernel(const float* __restrict__ pyr, float* __restrict__ out,
                             int B, int G)
{
    int g3 = G * G * G;
    int rowcnt = g3 * 3;
    int total = B * rowcnt * 3;
    int idx = blockIdx.x * blockDim.x + threadIdx.x;
    if (idx >= total) return;
    int b = (idx >= rowcnt * 3) ? 1 : 0;           // B == 2
    int rem = idx - b * rowcnt * 3;
    int r = rem / 3, t = rem - r * 3;
    int flat = r * 4 + t;                          // index into (G^3, 12)
    int vox = flat / 12, c = flat - vox * 12;
    int z = vox / (G * G), y = (vox / G) % G, x = vox % G;
    float f = pyr[(b * 12 + c) * g3 + vox];
    int j = c % 3;
    float cen = (j == 0) ? (x * 0.05f - 3.f)
              : (j == 1) ? (y * 0.05f - 3.f)
                         : (z * 0.05f - 1.f);
    out[idx] = (f == 0.f) ? 0.f : (cen + f * 0.05f);
}

// ============================================================
// host-side launch helpers
// ============================================================
static inline dim3 ew(int n) { return dim3((n + 255) / 256); }
static inline int ilog2i(int v) { int l = 0; while ((1 << l) < v) ++l; return l; }

static void launch_conv(int mode, const float* src, const float* wgt,
                        float* dst, _Float16* wrepack,
                        int B, int Cin, int Cout, int Gi, int Go, hipStream_t s)
{
    int taps  = (mode == CONV_1X1) ? 1 : 27;
    int lgCin = ilog2i(Cin), lgGi = ilog2i(Gi), lgGo = ilog2i(Go);
    int Kpad  = ((taps << lgCin) + 31) & ~31;
    int Coutp = (Cout + 15) & ~15;
    int M = B * Go * Go * Go;

    repack_w_kernel<<<ew(Coutp * Kpad), 256, 0, s>>>(wgt, wrepack, taps, Cin, lgCin,
                                                     Cout, Kpad, Coutp);
    dim3 grid(M / 128, Coutp / 16);
    dim3 blk(256);
    switch (mode) {
    case CONV_S2:
        wmma_conv_kernel<CONV_S2><<<grid, blk, 0, s>>>(src, wrepack, dst,
            B, Cin, lgCin, Cout, Gi, lgGi, Go, lgGo, Kpad);
        break;
    case CONV_1X1:
        wmma_conv_kernel<CONV_1X1><<<grid, blk, 0, s>>>(src, wrepack, dst,
            B, Cin, lgCin, Cout, Gi, lgGi, Go, lgGo, Kpad);
        break;
    default:
        wmma_conv_kernel<CONVT_S2><<<grid, blk, 0, s>>>(src, wrepack, dst,
            B, Cin, lgCin, Cout, Gi, lgGi, Go, lgGo, Kpad);
        break;
    }
}

extern "C" void kernel_launch(void* const* d_in, const int* in_sizes, int n_in,
                              void* d_out, int out_size, void* d_ws, size_t ws_size,
                              hipStream_t stream)
{
    (void)in_sizes; (void)n_in; (void)out_size; (void)ws_size;
    const int B = 2, G = 64;
    const int enc_in[4]  = {12, 16, 32, 64};
    const int enc_out[4] = {16, 32, 64, 128};
    const int dec_in[4]  = {16, 32, 64, 128};
    const int dec_out[4] = {12, 16, 32, 64};

    // ---------------- inputs ----------------
    const float* feats = (const float*)d_in[0];
    const float* occ   = (const float*)d_in[1];
    int pi = 2;
    const float *enc_w[4], *enc_g[4], *enc_b[4];
    for (int i = 0; i < 4; ++i) {
        enc_w[i] = (const float*)d_in[pi++];
        enc_g[i] = (const float*)d_in[pi++];
        enc_b[i] = (const float*)d_in[pi++];
    }
    const float *dec_w[4], *dec_g[4] = {0}, *dec_b[4] = {0};
    const float *cf_w[4], *cf_g[4], *cf_b[4], *occu_w[4], *occu_b[4];
    for (int l = 0; l < 4; ++l) {
        dec_w[l] = (const float*)d_in[pi++];
        if (l > 0) { dec_g[l] = (const float*)d_in[pi++];
                     dec_b[l] = (const float*)d_in[pi++]; }
        cf_w[l]   = (const float*)d_in[pi++];
        cf_g[l]   = (const float*)d_in[pi++];
        cf_b[l]   = (const float*)d_in[pi++];
        occu_w[l] = (const float*)d_in[pi++];
        occu_b[l] = (const float*)d_in[pi++];
    }
    const float* final_w = (const float*)d_in[pi++];
    const float* final_b = (const float*)d_in[pi++];

    // ---------------- workspace arena ----------------
    char* wp = (char*)d_ws;
    auto alloc = [&](size_t nfloats) -> float* {
        float* p = (float*)wp;
        wp += ((nfloats * sizeof(float)) + 255) & ~(size_t)255;
        return p;
    };
    const int S64 = G * G * G;
    float* x0 = alloc((size_t)B * 12 * S64);
    float* occl[4];
    float* eout[4];
    int gs[5] = {64, 32, 16, 8, 4};
    for (int i = 0; i < 4; ++i) {
        int Gd = gs[i + 1];
        occl[i] = alloc((size_t)B * Gd * Gd * Gd);
        eout[i] = alloc((size_t)B * enc_out[i] * Gd * Gd * Gd);
    }
    float* stats = alloc(256);
    float* cfb[3];  float* cob[3];
    for (int l = 0; l < 3; ++l) {
        int Gd = gs[l + 1];
        cfb[l] = alloc((size_t)B * dec_in[l] * Gd * Gd * Gd);
        cob[l] = alloc((size_t)B * Gd * Gd * Gd);
    }
    float* feat    = alloc((size_t)B * 128 * 32 * 32 * 32);
    float* probtmp = alloc((size_t)B * 32 * 32 * 32);
    float* keepbuf = alloc((size_t)B * 32 * 32 * 32);
    float* cfmask  = alloc((size_t)B * 16 * 32 * 32 * 32);   // cf*keep (max level 0)
    float* upA = alloc((size_t)B * 12 * S64);
    float* upB = alloc((size_t)B * 12 * S64);
    float* poA = alloc((size_t)B * S64);
    float* poB = alloc((size_t)B * S64);
    _Float16* wrepack = (_Float16*)alloc(128 * 3456 / 2 + 64);   // max Coutp*Kpad halves

    // ---------------- output offsets ----------------
    float* out_f = (float*)d_out;
    size_t off = 0;
    float* preds_o = out_f + off; off += (size_t)B * S64 * 3 * 3;
    float* cls_o[4]; float* tgt_o[4];
    int clsS[4] = {B * 32 * 32 * 32, B * 16 * 16 * 16, B * 8 * 8 * 8, B * 4 * 4 * 4};
    for (int l = 0; l < 4; ++l) { cls_o[l] = out_f + off; off += clsS[l]; }
    for (int l = 0; l < 4; ++l) { tgt_o[l] = out_f + off; off += clsS[l]; }
    float* dec_o = out_f + off;

    // ================= encoder =================
    mul_mask_kernel<<<ew(B * 12 * S64), 256, 0, stream>>>(x0, feats, occ, 12, B, 18);
    const float* x = x0;
    const float* co_cur = occ;
    for (int i = 0; i < 4; ++i) {
        int Gi = gs[i], Go = gs[i + 1];
        int lgGi = ilog2i(Gi), lgGo = ilog2i(Go);
        int S = Go * Go * Go;
        occ_down_kernel<<<ew(B * S), 256, 0, stream>>>(co_cur, occl[i], B, Gi, lgGi, Go, lgGo);
        launch_conv(CONV_S2, x, enc_w[i], eout[i], wrepack,
                    B, enc_in[i], enc_out[i], Gi, Go, stream);
        bn_stats_kernel<<<enc_out[i], 256, 0, stream>>>(eout[i], stats, enc_out[i], B, S);
        bn_apply_kernel<<<ew(B * enc_out[i] * S), 256, 0, stream>>>(
            eout[i], stats, enc_g[i], enc_b[i], occl[i], enc_out[i], B, 3 * lgGo);
        x = eout[i]; co_cur = occl[i];
    }

    // ================= decoder pyramid (coarse -> fine) =================
    float* pyr = nullptr; float* pyro = nullptr;
    int ping = 0;
    for (int l = 3; l >= 0; --l) {
        int Gd = gs[l + 1], S = Gd * Gd * Gd, Cl = dec_in[l];
        int lgGd = ilog2i(Gd);
        const float* cf; const float* co;
        if (pyr) {
            add_kernel<<<ew(B * Cl * S), 256, 0, stream>>>(cfb[l], eout[l], pyr, B * Cl * S);
            max_kernel<<<ew(B * S), 256, 0, stream>>>(cob[l], occl[l], pyro, B * S);
            cf = cfb[l]; co = cob[l];
        } else { cf = eout[3]; co = occl[3]; }

        // feat = relu(bn(1x1 conv -> 128)) * co
        launch_conv(CONV_1X1, cf, cf_w[l], feat, wrepack, B, Cl, 128, Gd, Gd, stream);
        bn_stats_kernel<<<128, 256, 0, stream>>>(feat, stats, 128, B, S);
        bn_apply_kernel<<<ew(B * 128 * S), 256, 0, stream>>>(
            feat, stats, cf_g[l], cf_b[l], co, 128, B, 3 * lgGd);

        // prob / tgt / keep
        launch_conv(CONV_1X1, feat, occu_w[l], probtmp, wrepack, B, 128, 1, Gd, Gd, stream);
        sigmoid_bias_kernel<<<ew(B * S), 256, 0, stream>>>(cls_o[l], probtmp, occu_b[l], B * S);
        gt0_kernel<<<ew(B * S), 256, 0, stream>>>(tgt_o[l], co, B * S);
        keep_kernel<<<ew(B * S), 256, 0, stream>>>(keepbuf, cls_o[l], co, B * S);

        // up = convT(cf * keep)  and  pyr_occ = occ_up(keep)
        int Gf = 2 * Gd, Sf = Gf * Gf * Gf, Cout = dec_out[l];
        int lgGf = lgGd + 1;
        float* up = ping ? upB : upA;
        float* po = ping ? poB : poA;
        mul_mask_kernel<<<ew(B * Cl * S), 256, 0, stream>>>(cfmask, cf, keepbuf, Cl, B, 3 * lgGd);
        launch_conv(CONVT_S2, cfmask, dec_w[l], up, wrepack, B, Cl, Cout, Gd, Gf, stream);
        occ_up_kernel<<<ew(B * Sf), 256, 0, stream>>>(keepbuf, po, B, Gd, lgGd, Gf, lgGf);
        if (l > 0) {
            bn_stats_kernel<<<Cout, 256, 0, stream>>>(up, stats, Cout, B, Sf);
            bn_apply_kernel<<<ew(B * Cout * Sf), 256, 0, stream>>>(
                up, stats, dec_g[l], dec_b[l], po, Cout, B, 3 * lgGf);
        } else {
            sigmoid_mask_kernel<<<ew(B * Cout * Sf), 256, 0, stream>>>(up, po, Cout, B, 3 * lgGf);
        }
        pyr = up; pyro = po; ping ^= 1;
    }

    // ================= postprocess =================
    preds_kernel<<<ew(B * S64 * 9), 256, 0, stream>>>(pyr, preds_o, B, G);
    launch_conv(CONV_1X1, pyr, final_w, dec_o, wrepack, B, 12, 1, G, G, stream);
    addbias_kernel<<<ew(B * S64), 256, 0, stream>>>(dec_o, final_b, B * S64);
}
